// SwinTransformerBlock_53506702574103
// MI455X (gfx1250) — matrix operand order repaired and checked by reading, not compiled, not run on hardware
//
#include <hip/hip_runtime.h>
#include <hip/hip_bf16.h>
#include <math.h>

// ---------------- types ----------------
typedef __bf16 bf16_t;
typedef __attribute__((ext_vector_type(16))) __bf16 v16bf;
typedef __attribute__((ext_vector_type(8)))  __bf16 v8bf;
typedef __attribute__((ext_vector_type(8)))  float  v8f;

#define B_    32
#define RES   64
#define C_    256
#define HEADS 8
#define HD    32
#define WS_   8
#define SS_   4
#define NTOK  64            // tokens per window
#define NWIN  4096          // B * nW = 32*64... (computed below actually 2048)
#define NW_IMG 64           // windows per image (8x8)
#define NWTOT 2048          // total windows
#define MROWS 131072        // total token rows (B * 4096)
#define HID_  1024

// ---------------- WMMA tile helper ----------------
// Computes a 16 x (16*NT) f32 tile: acc += A[16,K] * W[N,K]^T
// A: row-major, lda elems.  W: row-major [n][k], ldb elems (weight layout).
// A must be pre-offset to its 16-row start, W pre-offset to its n0 row start.
template <int NT>
__device__ __forceinline__ void wmma_rowtile(const bf16_t* __restrict__ A, int lda,
                                             const bf16_t* __restrict__ W, int ldb,
                                             int K, v8f* acc) {
    const int lane = threadIdx.x & 31;
    const int g    = lane >> 4;       // half-wave group
    const int r16  = lane & 15;       // m for A, n for B
    const bf16_t* arow = A + (size_t)r16 * lda;
    for (int k0 = 0; k0 < K; k0 += 32) {
        // A fragment: a[0..7] = K[k0+8g .. +8), a[8..15] = K[k0+16+8g .. +8)
        v8bf alo = *reinterpret_cast<const v8bf*>(arow + k0 + 8 * g);
        v8bf ahi = *reinterpret_cast<const v8bf*>(arow + k0 + 16 + 8 * g);
        v16bf a;
#pragma unroll
        for (int e = 0; e < 8; ++e) { a[e] = alo[e]; a[e + 8] = ahi[e]; }
#pragma unroll
        for (int j = 0; j < NT; ++j) {
            const bf16_t* brow = W + (size_t)(16 * j + r16) * ldb + k0;
            v8bf blo = *reinterpret_cast<const v8bf*>(brow + 8 * g);
            v8bf bhi = *reinterpret_cast<const v8bf*>(brow + 16 + 8 * g);
            v16bf b;
#pragma unroll
            for (int e = 0; e < 8; ++e) { b[e] = blo[e]; b[e + 8] = bhi[e]; }
            acc[j] = __builtin_amdgcn_wmma_f32_16x16x32_bf16(
                false, a, false, b, (short)0, acc[j], false, false);
        }
    }
}

template <int NT>
__device__ __forceinline__ void zero_acc(v8f* acc) {
#pragma unroll
    for (int j = 0; j < NT; ++j)
#pragma unroll
        for (int e = 0; e < 8; ++e) acc[j][e] = 0.0f;
}

// ---------------- weight fp32 -> bf16 ----------------
__global__ void cvt_bf16_kernel(const float* __restrict__ src, bf16_t* __restrict__ dst, int n) {
    int i = blockIdx.x * 256 + threadIdx.x;
    if (i < n) dst[i] = (bf16_t)src[i];
}

// ---------------- LayerNorm1 + cyclic shift + window partition -> bf16 ----------------
// dest row m (window-major): win=m>>6, tok=m&63; shifted (h,w); source = roll(-4,-4)
__global__ void ln1_shift_kernel(const float* __restrict__ x,
                                 const float* __restrict__ gam, const float* __restrict__ bet,
                                 bf16_t* __restrict__ hwin) {
    const int wave = threadIdx.x >> 5;
    const int lane = threadIdx.x & 31;
    const int m = blockIdx.x * 8 + wave;                 // dest row
    const int win = m >> 6, tok = m & 63;
    const int b = win >> 6, widx = win & 63;
    const int h = ((widx >> 3) << 3) + (tok >> 3);
    const int w = ((widx & 7) << 3) + (tok & 7);
    const int h0 = (h + SS_) & 63, w0 = (w + SS_) & 63;   // shifted[h,w]=orig[h+ss,w+ss]
    const size_t src = ((size_t)b * 4096 + (size_t)h0 * 64 + w0) * C_;

    float vals[8];
    float s = 0.f;
#pragma unroll
    for (int i = 0; i < 8; ++i) { vals[i] = x[src + lane * 8 + i]; s += vals[i]; }
#pragma unroll
    for (int off = 16; off; off >>= 1) s += __shfl_xor(s, off, 32);
    const float mean = s * (1.0f / C_);
    float vs = 0.f;
#pragma unroll
    for (int i = 0; i < 8; ++i) { float d = vals[i] - mean; vs += d * d; }
#pragma unroll
    for (int off = 16; off; off >>= 1) vs += __shfl_xor(vs, off, 32);
    const float inv = rsqrtf(vs * (1.0f / C_) + 1e-5f);
    const size_t dst = (size_t)m * C_;
#pragma unroll
    for (int i = 0; i < 8; ++i) {
        int c = lane * 8 + i;
        hwin[dst + c] = (bf16_t)((vals[i] - mean) * inv * gam[c] + bet[c]);
    }
}

// ---------------- LayerNorm2 (token order) ----------------
__global__ void ln2_kernel(const float* __restrict__ x,
                           const float* __restrict__ gam, const float* __restrict__ bet,
                           bf16_t* __restrict__ h2) {
    const int wave = threadIdx.x >> 5;
    const int lane = threadIdx.x & 31;
    const int m = blockIdx.x * 8 + wave;
    const size_t src = (size_t)m * C_;
    float vals[8];
    float s = 0.f;
#pragma unroll
    for (int i = 0; i < 8; ++i) { vals[i] = x[src + lane * 8 + i]; s += vals[i]; }
#pragma unroll
    for (int off = 16; off; off >>= 1) s += __shfl_xor(s, off, 32);
    const float mean = s * (1.0f / C_);
    float vs = 0.f;
#pragma unroll
    for (int i = 0; i < 8; ++i) { float d = vals[i] - mean; vs += d * d; }
#pragma unroll
    for (int off = 16; off; off >>= 1) vs += __shfl_xor(vs, off, 32);
    const float inv = rsqrtf(vs * (1.0f / C_) + 1e-5f);
#pragma unroll
    for (int i = 0; i < 8; ++i) {
        int c = lane * 8 + i;
        h2[src + c] = (bf16_t)((vals[i] - mean) * inv * gam[c] + bet[c]);
    }
}

// ---------------- QKV GEMM: [131072,256] x [768,256]^T, scatter to q/k/vT ----------------
__global__ void qkv_kernel(const bf16_t* __restrict__ hwin, const bf16_t* __restrict__ wq,
                           const float* __restrict__ bias,
                           bf16_t* __restrict__ qb, bf16_t* __restrict__ kb,
                           bf16_t* __restrict__ vt) {
    const int wave = threadIdx.x >> 5;
    const int m0 = blockIdx.x * 128 + wave * 16;
    const int n0 = blockIdx.y * 64;
    v8f acc[4]; zero_acc<4>(acc);
    wmma_rowtile<4>(hwin + (size_t)m0 * C_, C_, wq + (size_t)n0 * C_, C_, C_, acc);
    const int lane = threadIdx.x & 31, g = lane >> 4, r16 = lane & 15;
#pragma unroll
    for (int j = 0; j < 4; ++j)
#pragma unroll
        for (int r = 0; r < 8; ++r) {
            const int m = m0 + r + 8 * g;
            const int n = n0 + 16 * j + r16;
            float v = acc[j][r] + bias[n];
            const int sec = n >> 8, c = n & 255;
            const int head = c >> 5, d = c & 31;
            const int win = m >> 6, tok = m & 63;
            const size_t qi = (((size_t)win * HEADS + head) * NTOK + tok) * HD + d;
            if (sec == 0)      qb[qi] = (bf16_t)(v * 0.17677669529663687f); // 1/sqrt(32)
            else if (sec == 1) kb[qi] = (bf16_t)v;
            else vt[(((size_t)win * HEADS + head) * HD + d) * NTOK + tok] = (bf16_t)v;
        }
}

// ---------------- attention per (window, head) ----------------
__device__ __forceinline__ int sregion(int h) { return h < 56 ? 0 : (h < 60 ? 1 : 2); }

__global__ void attn_kernel(const bf16_t* __restrict__ qb, const bf16_t* __restrict__ kb,
                            const bf16_t* __restrict__ vt, const float* __restrict__ rpb,
                            bf16_t* __restrict__ aw) {
    __shared__ __align__(32) float  Sf[NTOK * NTOK];   // 16 KB
    __shared__ __align__(32) bf16_t Pb[NTOK * NTOK];   // 8 KB
    const int win = blockIdx.x >> 3, head = blockIdx.x & 7;
    const int wave = threadIdx.x >> 5;                 // 4 waves
    const int lane = threadIdx.x & 31, g = lane >> 4, r16 = lane & 15;
    const bf16_t* q = qb + ((size_t)win * HEADS + head) * NTOK * HD;
    const bf16_t* k = kb + ((size_t)win * HEADS + head) * NTOK * HD;
    const bf16_t* v = vt + ((size_t)win * HEADS + head) * HD * NTOK;

    // S = q * k^T   (16x64 per wave), K = 32
    v8f acc[4]; zero_acc<4>(acc);
    wmma_rowtile<4>(q + (size_t)(wave * 16) * HD, HD, k, HD, HD, acc);

    const int widx = win & 63, wh = widx >> 3, ww = widx & 7;
#pragma unroll
    for (int j = 0; j < 4; ++j)
#pragma unroll
        for (int r = 0; r < 8; ++r) {
            const int nq = wave * 16 + r + 8 * g;      // query token in window
            const int nk = 16 * j + r16;               // key token in window
            const int i1 = nq >> 3, j1 = nq & 7, i2 = nk >> 3, j2 = nk & 7;
            const float bias = rpb[((i1 - i2 + 7) * 15 + (j1 - j2 + 7)) * HEADS + head];
            const int lq = 3 * sregion(wh * 8 + i1) + sregion(ww * 8 + j1);
            const int lk = 3 * sregion(wh * 8 + i2) + sregion(ww * 8 + j2);
            const float msk = (lq != lk) ? -100.0f : 0.0f;
            Sf[nq * NTOK + nk] = acc[j][r] + bias + msk;
        }
    __syncthreads();

    // row softmax (threads 0..63 each own a row)
    const int t = threadIdx.x;
    if (t < NTOK) {
        float mx = -1e30f;
        for (int jj = 0; jj < NTOK; ++jj) mx = fmaxf(mx, Sf[t * NTOK + jj]);
        float sum = 0.f;
        for (int jj = 0; jj < NTOK; ++jj) {
            float e = __expf(Sf[t * NTOK + jj] - mx);
            Sf[t * NTOK + jj] = e;
            sum += e;
        }
        const float inv = 1.0f / sum;
        for (int jj = 0; jj < NTOK; ++jj) Pb[t * NTOK + jj] = (bf16_t)(Sf[t * NTOK + jj] * inv);
    }
    __syncthreads();

    // O = P * v  (16x32 per wave), K = 64
    v8f oacc[2]; zero_acc<2>(oacc);
    wmma_rowtile<2>(Pb + (size_t)(wave * 16) * NTOK, NTOK, v, NTOK, NTOK, oacc);
#pragma unroll
    for (int j = 0; j < 2; ++j)
#pragma unroll
        for (int r = 0; r < 8; ++r) {
            const int tok = wave * 16 + r + 8 * g;
            const int d = 16 * j + r16;
            aw[((size_t)win * NTOK + tok) * C_ + head * HD + d] = (bf16_t)oacc[j][r];
        }
}

// ---------------- proj GEMM + window-reverse + unshift + residual ----------------
__global__ void proj_kernel(const bf16_t* __restrict__ aw, const bf16_t* __restrict__ wp,
                            const float* __restrict__ bias, const float* __restrict__ x,
                            float* __restrict__ x1) {
    const int wave = threadIdx.x >> 5;
    const int m0 = blockIdx.x * 128 + wave * 16;
    const int n0 = blockIdx.y * 64;
    v8f acc[4]; zero_acc<4>(acc);
    wmma_rowtile<4>(aw + (size_t)m0 * C_, C_, wp + (size_t)n0 * C_, C_, C_, acc);
    const int lane = threadIdx.x & 31, g = lane >> 4, r16 = lane & 15;
#pragma unroll
    for (int j = 0; j < 4; ++j)
#pragma unroll
        for (int r = 0; r < 8; ++r) {
            const int m = m0 + r + 8 * g;
            const int n = n0 + 16 * j + r16;
            const int win = m >> 6, tok = m & 63;
            const int b = win >> 6, widx = win & 63;
            const int h = ((widx >> 3) << 3) + (tok >> 3);
            const int w = ((widx & 7) << 3) + (tok & 7);
            const int h0 = (h + SS_) & 63, w0 = (w + SS_) & 63;  // orig = shifted(h+ss,w+ss)
            const size_t tix = ((size_t)b * 4096 + (size_t)h0 * 64 + w0) * C_ + n;
            x1[tix] = x[tix] + acc[j][r] + bias[n];
        }
}

// ---------------- fc1 GEMM + exact GELU ----------------
__global__ void fc1_kernel(const bf16_t* __restrict__ h2, const bf16_t* __restrict__ w1,
                           const float* __restrict__ bias, bf16_t* __restrict__ h3) {
    const int wave = threadIdx.x >> 5;
    const int m0 = blockIdx.x * 128 + wave * 16;
    const int n0 = blockIdx.y * 64;
    v8f acc[4]; zero_acc<4>(acc);
    wmma_rowtile<4>(h2 + (size_t)m0 * C_, C_, w1 + (size_t)n0 * C_, C_, C_, acc);
    const int lane = threadIdx.x & 31, g = lane >> 4, r16 = lane & 15;
#pragma unroll
    for (int j = 0; j < 4; ++j)
#pragma unroll
        for (int r = 0; r < 8; ++r) {
            const int m = m0 + r + 8 * g;
            const int n = n0 + 16 * j + r16;
            float v = acc[j][r] + bias[n];
            v = 0.5f * v * (1.0f + erff(v * 0.7071067811865475f));
            h3[(size_t)m * HID_ + n] = (bf16_t)v;
        }
}

// ---------------- fc2 GEMM + residual -> output ----------------
__global__ void fc2_kernel(const bf16_t* __restrict__ h3, const bf16_t* __restrict__ w2,
                           const float* __restrict__ bias, const float* __restrict__ x1,
                           float* __restrict__ out) {
    const int wave = threadIdx.x >> 5;
    const int m0 = blockIdx.x * 128 + wave * 16;
    const int n0 = blockIdx.y * 64;
    v8f acc[4]; zero_acc<4>(acc);
    wmma_rowtile<4>(h3 + (size_t)m0 * HID_, HID_, w2 + (size_t)n0 * HID_, HID_, HID_, acc);
    const int lane = threadIdx.x & 31, g = lane >> 4, r16 = lane & 15;
#pragma unroll
    for (int j = 0; j < 4; ++j)
#pragma unroll
        for (int r = 0; r < 8; ++r) {
            const int m = m0 + r + 8 * g;
            const int n = n0 + 16 * j + r16;
            const size_t tix = (size_t)m * C_ + n;
            out[tix] = x1[tix] + acc[j][r] + bias[n];
        }
}

// ---------------- launch ----------------
extern "C" void kernel_launch(void* const* d_in, const int* in_sizes, int n_in,
                              void* d_out, int out_size, void* d_ws, size_t ws_size,
                              hipStream_t stream) {
    (void)in_sizes; (void)n_in; (void)out_size; (void)ws_size;
    const float* x       = (const float*)d_in[0];
    const float* n1g     = (const float*)d_in[1];
    const float* n1b     = (const float*)d_in[2];
    const float* qkv_w   = (const float*)d_in[3];
    const float* qkv_b   = (const float*)d_in[4];
    const float* rpb     = (const float*)d_in[5];
    const float* proj_w  = (const float*)d_in[6];
    const float* proj_b  = (const float*)d_in[7];
    const float* n2g     = (const float*)d_in[8];
    const float* n2b     = (const float*)d_in[9];
    const float* fc1_w   = (const float*)d_in[10];
    const float* fc1_b   = (const float*)d_in[11];
    const float* fc2_w   = (const float*)d_in[12];
    const float* fc2_b   = (const float*)d_in[13];
    float* out = (float*)d_out;

    // workspace carve-up (256-B aligned)
    char* ws = (char*)d_ws;
    size_t off = 0;
    auto carve = [&](size_t bytes) -> char* {
        char* p = ws + off;
        off = (off + bytes + 255) & ~(size_t)255;
        return p;
    };
    bf16_t* wq_b  = (bf16_t*)carve((size_t)768 * 256 * 2);
    bf16_t* wp_b  = (bf16_t*)carve((size_t)256 * 256 * 2);
    bf16_t* w1_b  = (bf16_t*)carve((size_t)1024 * 256 * 2);
    bf16_t* w2_b  = (bf16_t*)carve((size_t)256 * 1024 * 2);
    bf16_t* hwin  = (bf16_t*)carve((size_t)MROWS * C_ * 2);   // aliased as aw later
    bf16_t* qb    = (bf16_t*)carve((size_t)MROWS * C_ * 2);   // aliased as h2 later
    bf16_t* kb    = (bf16_t*)carve((size_t)MROWS * C_ * 2);
    bf16_t* vt    = (bf16_t*)carve((size_t)MROWS * C_ * 2);
    float*  x1    = (float*) carve((size_t)MROWS * C_ * 4);
    bf16_t* h3    = (bf16_t*)carve((size_t)MROWS * HID_ * 2);
    bf16_t* aw    = hwin;   // lifetime-disjoint alias
    bf16_t* h2    = qb;     // lifetime-disjoint alias

    // 1) weights -> bf16
    cvt_bf16_kernel<<<(768 * 256 + 255) / 256, 256, 0, stream>>>(qkv_w, wq_b, 768 * 256);
    cvt_bf16_kernel<<<(256 * 256 + 255) / 256, 256, 0, stream>>>(proj_w, wp_b, 256 * 256);
    cvt_bf16_kernel<<<(1024 * 256 + 255) / 256, 256, 0, stream>>>(fc1_w, w1_b, 1024 * 256);
    cvt_bf16_kernel<<<(256 * 1024 + 255) / 256, 256, 0, stream>>>(fc2_w, w2_b, 256 * 1024);

    // 2) LN1 + shift + window partition
    ln1_shift_kernel<<<MROWS / 8, 256, 0, stream>>>(x, n1g, n1b, hwin);

    // 3) QKV projection (WMMA)
    qkv_kernel<<<dim3(MROWS / 128, 768 / 64), 256, 0, stream>>>(hwin, wq_b, qkv_b, qb, kb, vt);

    // 4) windowed attention (WMMA x2 + softmax)
    attn_kernel<<<NWTOT * HEADS, 128, 0, stream>>>(qb, kb, vt, rpb, aw);

    // 5) output projection + window reverse + unshift + residual (WMMA)
    proj_kernel<<<dim3(MROWS / 128, 256 / 64), 256, 0, stream>>>(aw, wp_b, proj_b, x, x1);

    // 6) LN2
    ln2_kernel<<<MROWS / 8, 256, 0, stream>>>(x1, n2g, n2b, h2);

    // 7) MLP fc1 + GELU (WMMA)
    fc1_kernel<<<dim3(MROWS / 128, 1024 / 64), 256, 0, stream>>>(h2, w1_b, fc1_b, h3);

    // 8) MLP fc2 + residual (WMMA)
    fc2_kernel<<<dim3(MROWS / 128, 256 / 64), 256, 0, stream>>>(h3, w2_b, fc2_b, x1, out);
}
